// Dual_axis_69922067578981
// MI455X (gfx1250) — matrix operand order repaired and checked by reading, not compiled, not run on hardware
//
#include <hip/hip_runtime.h>

// ---------- types ----------
typedef __attribute__((ext_vector_type(16))) __bf16 v16bf;
typedef __attribute__((ext_vector_type(8)))  __bf16 v8bf;
typedef __attribute__((ext_vector_type(8)))  float  v8f;
typedef int v4i __attribute__((__vector_size__(16)));   // matches builtin's V4i

#define B_  16
#define C_  256
#define S_  64
#define H_  8
#define D_  64
#define S2_ 4096   // S_*S_

#define GLOBAL_AS __attribute__((address_space(1)))
#define LDS_AS    __attribute__((address_space(3)))

#if defined(__has_builtin)
#if __has_builtin(__builtin_amdgcn_global_load_async_to_lds_b128) && \
    __has_builtin(__builtin_amdgcn_s_wait_asynccnt)
#define HAVE_ASYNC_LDS 1
#endif
#endif

static __device__ __forceinline__ __bf16 to_bf16(float f) { return (__bf16)f; }

// Cooperative copy of `nFloats` f32 from global `src` to LDS `dst` by `nthr` threads.
// Uses gfx1250 async global->LDS b128 when the toolchain exposes the builtin.
static __device__ __forceinline__ void stage_to_lds(const float* __restrict__ src,
                                                    float* __restrict__ dst,
                                                    int nFloats, int tid, int nthr) {
#if defined(HAVE_ASYNC_LDS)
    const unsigned long long gbase = (unsigned long long)(const void*)src;
    const unsigned           lbase = (unsigned)(unsigned long long)(const void*)dst;
    const int perIter = nthr * 4;                 // floats per pass (b128 per lane)
    for (int base = 0; base < nFloats; base += perIter) {
        const int idx = base + tid * 4;
        __builtin_amdgcn_global_load_async_to_lds_b128(
            (GLOBAL_AS v4i*)(gbase + (unsigned long long)idx * 4ull),
            (LDS_AS v4i*)(lbase + (unsigned)idx * 4u), 0, 0);
    }
    __builtin_amdgcn_s_wait_asynccnt(0);
#else
    for (int i = tid; i < nFloats; i += nthr) dst[i] = src[i];
#endif
}

// ---------- weight cast f32 -> bf16 ----------
__global__ void cast_bf16_kernel(const float* __restrict__ in, __bf16* __restrict__ out, int n) {
    int i = blockIdx.x * blockDim.x + threadIdx.x;
    if (i < n) out[i] = to_bf16(in[i]);
}

// ---------- depthwise 3x3 conv + mean over one spatial axis ----------
// One block per (b,c); 64 threads; output pooledT[(b*64 + l)*256 + c] in bf16.
// poolWidth==1 : mean over x (axis 3), l = y.  poolWidth==0 : mean over y, l = x.
__global__ __launch_bounds__(64)
void dwconv_pool_kernel(const float* __restrict__ x, const float* __restrict__ w,
                        const float* __restrict__ bdw, __bf16* __restrict__ outT,
                        int poolWidth) {
    __shared__ float plane[S_ * S_];
    const int bc = blockIdx.x;
    const int b  = bc >> 8;
    const int c  = bc & 255;
    const float* src = x + (long long)bc * S2_;
    stage_to_lds(src, plane, S2_, threadIdx.x, 64);
    __syncthreads();

    float wv[9];
#pragma unroll
    for (int t = 0; t < 9; ++t) wv[t] = w[c * 9 + t];
    const float bb = bdw[c];

    const int l = threadIdx.x;  // 0..63
    float s = 0.f;
    for (int t = 0; t < S_; ++t) {
        const int y  = poolWidth ? l : t;
        const int xx = poolWidth ? t : l;
        float acc = 0.f;
#pragma unroll
        for (int dy = -1; dy <= 1; ++dy) {
            const int yy = y + dy;
            if (yy < 0 || yy > 63) continue;
#pragma unroll
            for (int dx = -1; dx <= 1; ++dx) {
                const int xc = xx + dx;
                if (xc < 0 || xc > 63) continue;
                acc += plane[yy * S_ + xc] * wv[(dy + 1) * 3 + (dx + 1)];
            }
        }
        s += acc;
    }
    const float val = s * (1.0f / 64.0f) + bb;
    outT[((long long)(b * S_ + l)) * C_ + c] = to_bf16(val);
}

// ---------- depthwise 3x3 conv (full plane) -> bf16 activation [B*4096, 256] ----------
__global__ __launch_bounds__(256)
void dwconv_full_kernel(const float* __restrict__ x, const float* __restrict__ w,
                        const float* __restrict__ bdw, __bf16* __restrict__ Xv) {
    __shared__ float plane[S_ * S_];
    const int bc = blockIdx.x;
    const int b  = bc >> 8;
    const int c  = bc & 255;
    const float* src = x + (long long)bc * S2_;
    stage_to_lds(src, plane, S2_, threadIdx.x, 256);
    __syncthreads();

    float wv[9];
#pragma unroll
    for (int t = 0; t < 9; ++t) wv[t] = w[c * 9 + t];
    const float bb = bdw[c];

    for (int i = threadIdx.x; i < S2_; i += 256) {
        const int y  = i >> 6;
        const int xx = i & 63;
        float acc = 0.f;
#pragma unroll
        for (int dy = -1; dy <= 1; ++dy) {
            const int yy = y + dy;
            if (yy < 0 || yy > 63) continue;
#pragma unroll
            for (int dx = -1; dx <= 1; ++dx) {
                const int xc = xx + dx;
                if (xc < 0 || xc > 63) continue;
                acc += plane[yy * S_ + xc] * wv[(dy + 1) * 3 + (dx + 1)];
            }
        }
        Xv[((long long)(b * S2_ + i)) * C_ + c] = to_bf16(acc + bb);
    }
}

// ---------- WMMA bf16 GEMM: out = A[M,K] @ W[N,K]^T + bias ----------
// One wave computes a 16(M) x 64(N) tile: the A fragment is loaded once per
// k-step and reused by 4 WMMAs (4x less A traffic; B = weights stay L2-hot).
// CDNA5 ISA layouts:
//   A frag : lanes 0-15 row M=lane, K-base 0; lanes 16-31 same rows, K-base 8;
//            elements 0..7 = K+0..7, elements 8..15 = K+16..23.
//   B frag : same K order as A frag (lanes 0-15 col N=lane, K-base 0; lanes 16-31 K-base 8).
//   C/D    : VGPR r, N = lane&15, M = r + 8*(lane>=16).
// mode 0: f32 row-major out;  mode 1: bf16 vv[b,h,s,d];  mode 2: bf16 rfinal[b,s,h*64+n]
__global__ __launch_bounds__(128)
void gemm_bf16_wmma(const __bf16* __restrict__ A, long long aStride,
                    const __bf16* __restrict__ W, long long wStride,
                    const float* __restrict__ bias,
                    void* __restrict__ Out,
                    int M, int N, int K, int nBatch, int mode) {
    const int tN = N >> 6;                 // 64-wide N tiles
    const int tM = M >> 4;
    const int tilesPerBatch = tM * tN;
    const int wid = (int)((blockIdx.x * blockDim.x + threadIdx.x) >> 5);
    if (wid >= tilesPerBatch * nBatch) return;
    const int lane  = threadIdx.x & 31;
    const int batch = wid / tilesPerBatch;
    const int rem   = wid - batch * tilesPerBatch;
    const int tileM = rem / tN;
    const int tileN = rem - tileM * tN;

    const __bf16* Ab = A + (long long)batch * aStride;
    const __bf16* Wb = W + (long long)batch * wStride;

    const int hi   = lane >> 4;
    const int row  = (tileM << 4) + (lane & 15);
    const int col0 = (tileN << 6) + (lane & 15);
    const int aOff = hi << 3;   // K-base 0 / 8
    const int bOff = hi << 3;   // K-base 0 / 8

    v8f acc[4] = {v8f{}, v8f{}, v8f{}, v8f{}};
    for (int k = 0; k < K; k += 32) {
        const __bf16* ap = Ab + (long long)row * K + (k + aOff);
        v8bf a0 = *reinterpret_cast<const v8bf*>(ap);
        v8bf a1 = *reinterpret_cast<const v8bf*>(ap + 16);
        v16bf afrag = __builtin_shufflevector(a0, a1, 0, 1, 2, 3, 4, 5, 6, 7,
                                              8, 9, 10, 11, 12, 13, 14, 15);
#pragma unroll
        for (int j = 0; j < 4; ++j) {
            const __bf16* bp = Wb + (long long)(col0 + j * 16) * K + (k + bOff);
            v8bf b0 = *reinterpret_cast<const v8bf*>(bp);
            v8bf b1 = *reinterpret_cast<const v8bf*>(bp + 16);
            v16bf bfrag = __builtin_shufflevector(b0, b1, 0, 1, 2, 3, 4, 5, 6, 7,
                                                  8, 9, 10, 11, 12, 13, 14, 15);
            acc[j] = __builtin_amdgcn_wmma_f32_16x16x32_bf16(false, afrag, false, bfrag,
                                                             (short)0, acc[j], false, false);
        }
    }

    const int nLoc = lane & 15;
#pragma unroll
    for (int j = 0; j < 4; ++j) {
#pragma unroll
        for (int r = 0; r < 8; ++r) {
            const int m = (tileM << 4) + r + (hi << 3);
            const int n = (tileN << 6) + j * 16 + nLoc;
            float v = acc[j][r] + (bias ? bias[n] : 0.0f);
            if (mode == 0) {
                reinterpret_cast<float*>(Out)[(long long)batch * M * N + (long long)m * N + n] = v;
            } else if (mode == 1) {  // vv[b,h,s,d], m = b*4096+s, n = h*64+d
                const int b = m >> 12, s = m & 4095, h = n >> 6, d = n & 63;
                reinterpret_cast<__bf16*>(Out)[((((long long)(b * H_ + h)) * S2_ + s) << 6) + d] =
                    to_bf16(v);
            } else {                 // rfinal[b, s, h*64+n], batch = b*8+h, m = s
                const int b = batch >> 3, h = batch & 7;
                reinterpret_cast<__bf16*>(Out)[((long long)(b * S2_ + m)) * 512 + (h << 6) + n] =
                    to_bf16(v);
            }
        }
    }
}

// ---------- L2 normalize over L and transpose: P[b,l,j] -> Qn[b,h,d,l] ----------
__global__ void l2norm_kernel(const float* __restrict__ P, float* __restrict__ Qn) {
    const int idx = blockIdx.x * blockDim.x + threadIdx.x;  // b*512 + j
    if (idx >= B_ * 512) return;
    const int b = idx >> 9, j = idx & 511;
    const float* p = P + (long long)b * S_ * 512 + j;
    float ss = 0.f;
    for (int l = 0; l < S_; ++l) { float t = p[l * 512]; ss += t * t; }
    const float inv = 1.0f / fmaxf(sqrtf(ss), 1e-12f);
    float* qo = Qn + (((long long)(b * H_ + (j >> 6)) * D_) + (j & 63)) * S_;
    for (int l = 0; l < S_; ++l) qo[l] = p[l * 512] * inv;
}

// ---------- per-(b,h) scores + softmax: S[d,e]=sum_l Q[d,l]K[e,l]; softmax over e ----------
__global__ __launch_bounds__(64)
void attn_softmax_kernel(const float* __restrict__ Qn, const float* __restrict__ Kn,
                         float* __restrict__ attn) {
    __shared__ float Qs[D_ * S_];
    __shared__ float Ks[D_ * S_];
    __shared__ float Sc[D_ * D_];
    const int bh = blockIdx.x;
    stage_to_lds(Qn + (long long)bh * 4096, Qs, 4096, threadIdx.x, 64);
    stage_to_lds(Kn + (long long)bh * 4096, Ks, 4096, threadIdx.x, 64);
    __syncthreads();

    const int d = threadIdx.x;
    float mx = -3.4e38f;
    for (int e = 0; e < D_; ++e) {
        float dot = 0.f;
        for (int l = 0; l < S_; ++l) dot += Qs[d * S_ + l] * Ks[e * S_ + l];
        Sc[d * D_ + e] = dot;
        mx = fmaxf(mx, dot);
    }
    float sum = 0.f;
    for (int e = 0; e < D_; ++e) {
        float ex = __expf(Sc[d * D_ + e] - mx);
        Sc[d * D_ + e] = ex;
        sum += ex;
    }
    const float inv = 1.0f / sum;
    float* out = attn + (long long)bh * 4096;
    for (int e = 0; e < D_; ++e) out[d * D_ + e] = Sc[d * D_ + e] * inv;
}

// ---------- combine: Wap[bh][e][d] = sum_g attn_h[g,d] * attn_w[g,e]  (bf16) ----------
__global__ __launch_bounds__(256)
void combine_attn_kernel(const float* __restrict__ Ah, const float* __restrict__ Aw,
                         __bf16* __restrict__ Wap) {
    __shared__ float Hs[4096];
    __shared__ float Ws[4096];
    const int bh = blockIdx.x;
    stage_to_lds(Ah + (long long)bh * 4096, Hs, 4096, threadIdx.x, 256);
    stage_to_lds(Aw + (long long)bh * 4096, Ws, 4096, threadIdx.x, 256);
    __syncthreads();
    for (int i = threadIdx.x; i < 4096; i += 256) {
        const int e = i >> 6, d = i & 63;
        float s = 0.f;
        for (int g = 0; g < D_; ++g) s += Hs[g * D_ + d] * Ws[g * D_ + e];
        Wap[(long long)bh * 4096 + i] = to_bf16(s);
    }
}

// ---------- host ----------
extern "C" void kernel_launch(void* const* d_in, const int* in_sizes, int n_in,
                              void* d_out, int out_size, void* d_ws, size_t ws_size,
                              hipStream_t stream) {
    const float* q = (const float*)d_in[0];
    const float* k = (const float*)d_in[1];
    const float* v = (const float*)d_in[2];
    const float* wdw_qh = (const float*)d_in[3];  const float* bdw_qh = (const float*)d_in[4];
    const float* W_qh   = (const float*)d_in[5];  const float* b_qh   = (const float*)d_in[6];
    const float* wdw_kh = (const float*)d_in[7];  const float* bdw_kh = (const float*)d_in[8];
    const float* W_kh   = (const float*)d_in[9];  const float* b_kh   = (const float*)d_in[10];
    const float* wdw_v  = (const float*)d_in[11]; const float* bdw_v  = (const float*)d_in[12];
    const float* W_v    = (const float*)d_in[13]; const float* b_v    = (const float*)d_in[14];
    const float* wdw_qw = (const float*)d_in[15]; const float* bdw_qw = (const float*)d_in[16];
    const float* W_qw   = (const float*)d_in[17]; const float* b_qw   = (const float*)d_in[18];
    const float* wdw_kw = (const float*)d_in[19]; const float* bdw_kw = (const float*)d_in[20];
    const float* W_kw   = (const float*)d_in[21]; const float* b_kw   = (const float*)d_in[22];
    const float* W_o    = (const float*)d_in[23]; const float* b_o    = (const float*)d_in[24];

    char* ws = (char*)d_ws;
    size_t off = 0;
    auto take = [&](size_t bytes) -> void* {
        void* p = ws + off;
        off = (off + bytes + 255) & ~(size_t)255;
        return p;
    };

    __bf16* pooled[4];
    for (int i = 0; i < 4; ++i) pooled[i] = (__bf16*)take((size_t)1024 * 256 * 2);
    float* P[4];
    for (int i = 0; i < 4; ++i) P[i] = (float*)take((size_t)1024 * 512 * 4);
    float* Qn[4];
    for (int i = 0; i < 4; ++i) Qn[i] = (float*)take((size_t)B_ * H_ * D_ * S_ * 4);
    float* attn_h = (float*)take((size_t)128 * 4096 * 4);
    float* attn_w = (float*)take((size_t)128 * 4096 * 4);
    __bf16* Wap = (__bf16*)take((size_t)128 * 4096 * 2);
    __bf16* Xv  = (__bf16*)take((size_t)65536 * 256 * 2);
    __bf16* vv  = (__bf16*)take((size_t)B_ * H_ * S2_ * D_ * 2);
    __bf16* rfin = (__bf16*)take((size_t)65536 * 512 * 2);
    __bf16* Wbf[6];
    for (int i = 0; i < 6; ++i) Wbf[i] = (__bf16*)take((size_t)131072 * 2);
    (void)ws_size;

    // 1) cast linear weights to bf16: {W_qh, W_kh, W_v, W_qw, W_kw, W_o}
    {
        const float* srcs[6] = {W_qh, W_kh, W_v, W_qw, W_kw, W_o};
        for (int i = 0; i < 6; ++i)
            cast_bf16_kernel<<<(131072 + 255) / 256, 256, 0, stream>>>(srcs[i], Wbf[i], 131072);
    }

    // 2) dwconv + axis pooling for the four q/k branches
    dwconv_pool_kernel<<<B_ * C_, 64, 0, stream>>>(q, wdw_qh, bdw_qh, pooled[0], 1);
    dwconv_pool_kernel<<<B_ * C_, 64, 0, stream>>>(k, wdw_kh, bdw_kh, pooled[1], 1);
    dwconv_pool_kernel<<<B_ * C_, 64, 0, stream>>>(q, wdw_qw, bdw_qw, pooled[2], 0);
    dwconv_pool_kernel<<<B_ * C_, 64, 0, stream>>>(k, wdw_kw, bdw_kw, pooled[3], 0);

    // 3) small projections: [1024,256] @ [512,256]^T -> f32 P
    {
        const float* biases[4] = {b_qh, b_kh, b_qw, b_kw};
        const int wI[4] = {0, 1, 3, 4};
        const int waves = (1024 / 16) * (512 / 64);
        for (int i = 0; i < 4; ++i)
            gemm_bf16_wmma<<<(waves + 3) / 4, 128, 0, stream>>>(
                pooled[i], 0, Wbf[wI[i]], 0, biases[i], P[i], 1024, 512, 256, 1, 0);
    }

    // 4) L2 normalize over L, transpose to [b,h,d,l]
    for (int i = 0; i < 4; ++i)
        l2norm_kernel<<<(B_ * 512 + 255) / 256, 256, 0, stream>>>(P[i], Qn[i]);

    // 5) attention scores + softmax; combine into Wap
    attn_softmax_kernel<<<128, 64, 0, stream>>>(Qn[0], Qn[1], attn_h);
    attn_softmax_kernel<<<128, 64, 0, stream>>>(Qn[2], Qn[3], attn_w);
    combine_attn_kernel<<<128, 256, 0, stream>>>(attn_h, attn_w, Wap);

    // 6) value path: dwconv -> Xv bf16; project -> vv[b,h,s,d] bf16
    dwconv_full_kernel<<<B_ * C_, 256, 0, stream>>>(v, wdw_v, bdw_v, Xv);
    {
        const int waves = (65536 / 16) * (512 / 64);
        gemm_bf16_wmma<<<(waves + 3) / 4, 128, 0, stream>>>(
            Xv, 0, Wbf[2], 0, b_v, vv, 65536, 512, 256, 1, 1);
    }

    // 7) apply combined attention: rfinal[b,s,512] = vv @ Wap (batched 128)
    {
        const int waves = 128 * (4096 / 16) * (64 / 64);
        gemm_bf16_wmma<<<(waves + 3) / 4, 128, 0, stream>>>(
            vv, (long long)S2_ * D_, Wap, 4096, nullptr, rfin, 4096, 64, 64, 128, 2);
    }

    // 8) output projection: [65536,512] @ [256,512]^T + b_o -> d_out f32
    {
        const int waves = (65536 / 16) * (256 / 64);
        gemm_bf16_wmma<<<(waves + 3) / 4, 128, 0, stream>>>(
            rfin, 0, Wbf[5], 0, b_o, (float*)d_out, 65536, 256, 512, 1, 0);
    }
}
